// DeformableCrossAttention_59854664237575
// MI455X (gfx1250) — matrix operand order repaired and checked by reading, not compiled, not run on hardware
//
#include <hip/hip_runtime.h>
#include <hip/hip_bf16.h>
#include <cstdint>

#define NHEADS 8
#define HDIM   32
#define CDIM   256
#define NLVL   3
#define NPTS   8
#define BATCH  4
#define NQ     2048
#define SLEN   13125

typedef __attribute__((ext_vector_type(16))) _Float16 v16h;
typedef __attribute__((ext_vector_type(8)))  float    v8f;

struct alignas(16) H8 { _Float16 h[8]; };

// ---------------- WMMA fragment loaders (layouts per CDNA5 ISA) ----

// A 16x32 f16: lane<16 -> row M=lane, K {0..7, 16..23}; lane>=16 -> row M=lane-16, K {8..15, 24..31}
__device__ __forceinline__ v16h load_a_frag(const _Float16* __restrict__ A, int ld,
                                            int row0, int maxrow, int k0, int lane) {
  int r = lane & 15, hi = lane >> 4;
  int row = row0 + r; if (row >= maxrow) row = maxrow - 1;
  const _Float16* p = A + (size_t)row * ld + k0 + hi * 8;
  const H8* p0 = (const H8*)p;
  const H8* p1 = (const H8*)(p + 16);
  v16h f;
#pragma unroll
  for (int i = 0; i < 8; ++i) { f[i] = p0->h[i]; f[i + 8] = p1->h[i]; }
  return f;
}

// B 32x16 f16 (BT stored [N][K] row-major): lane<16 -> col N=lane, K {0..7, 16..23}; lane>=16 -> col, K {8..15, 24..31}
__device__ __forceinline__ v16h load_b_frag(const _Float16* __restrict__ BT, int ld,
                                            int col0, int maxcol, int k0, int lane) {
  int n = lane & 15, hi = lane >> 4;
  int col = col0 + n; if (col >= maxcol) col = maxcol - 1;
  const _Float16* p = BT + (size_t)col * ld + k0 + hi * 8;
  const H8* p0 = (const H8*)p;
  const H8* p1 = (const H8*)(p + 16);
  v16h f;
#pragma unroll
  for (int i = 0; i < 8; ++i) { f[i] = p0->h[i]; f[i + 8] = p1->h[i]; }
  return f;
}

// ---------------- Generic WMMA GEMM: C = A(f16,[M,K]) x BT(f16,[N,K])^T + bias ---
// mode 0: row-major f32 out (optional f16 dup). mode 1: scatter to value layout [B][NH][S][HD].
__global__ void gemm_wmma(const _Float16* __restrict__ A, const _Float16* __restrict__ BT,
                          const float* __restrict__ bias, float* __restrict__ Cout,
                          _Float16* __restrict__ Cout16, int M, int N, int K, int mode) {
  int gwave = (int)((blockIdx.x * blockDim.x + threadIdx.x) >> 5);
  int lane = threadIdx.x & 31;
  int ntiles = N >> 4;
  int mtiles = (M + 15) >> 4;
  int mt = gwave / ntiles;
  int nt = gwave - mt * ntiles;
  if (mt >= mtiles) return;          // uniform per-wave exit: EXEC stays all-ones for WMMA
  v8f c = {};
  for (int k0 = 0; k0 < K; k0 += 32) {
    v16h a = load_a_frag(A, K, mt << 4, M, k0, lane);
    v16h b = load_b_frag(BT, K, nt << 4, N, k0, lane);
    c = __builtin_amdgcn_wmma_f32_16x16x32_f16(false, a, false, b, (short)0, c, false, false);
  }
  int col = (nt << 4) + (lane & 15);
  int hi = lane >> 4;
  float bv = bias[col];
#pragma unroll
  for (int i = 0; i < 8; ++i) {
    int row = (mt << 4) + hi * 8 + i;
    if (row < M) {
      float v = c[i] + bv;
      if (mode == 0) {
        Cout[(size_t)row * N + col] = v;
        if (Cout16) Cout16[(size_t)row * N + col] = (_Float16)v;
      } else {
        int b_ = row / SLEN, s = row - b_ * SLEN;
        int hh = col >> 5, d = col & 31;
        Cout[((((size_t)b_ * NHEADS + hh) * SLEN + s) << 5) + d] = v;
      }
    }
  }
}

// ---------------- Prep kernels -------------------------------------------------
__global__ void wt_transpose_h(const float* __restrict__ W, _Float16* __restrict__ WT,
                               int K, int N) {  // WT[n*K+k] = W[k*N+n]
  int idx = blockIdx.x * blockDim.x + threadIdx.x;
  if (idx >= K * N) return;
  int n = idx / K;
  int k = idx - n * K;
  WT[idx] = (_Float16)W[(size_t)k * N + n];
}

__global__ void f32_to_f16(const float* __restrict__ X, _Float16* __restrict__ Y, int n) {
  int i = blockIdx.x * blockDim.x + threadIdx.x;
  if (i < n) Y[i] = (_Float16)X[i];
}

// ---------------- Softmax over 24 + sampling-location precompute ---------------
__global__ void softmax_grid_kernel(const float* __restrict__ offbuf, const float* __restrict__ logits,
                                    const float* __restrict__ refpts, float* __restrict__ attnw,
                                    float* __restrict__ grid) {
  int t = blockIdx.x * blockDim.x + threadIdx.x;   // t = (b*Nq+q)*NH + h
  if (t >= BATCH * NQ * NHEADS) return;
  int h = t & (NHEADS - 1);
  int bq = t >> 3;
  const float* lg = logits + (size_t)bq * (NHEADS * NLVL * NPTS) + h * (NLVL * NPTS);
  float m = -1e30f;
#pragma unroll
  for (int i = 0; i < 24; ++i) m = fmaxf(m, lg[i]);
  float e[24]; float s = 0.f;
#pragma unroll
  for (int i = 0; i < 24; ++i) { e[i] = __expf(lg[i] - m); s += e[i]; }
  float inv = 1.0f / s;
  float* aw = attnw + (size_t)t * 24;
#pragma unroll
  for (int i = 0; i < 24; ++i) aw[i] = e[i] * inv;
  float rx = refpts[(size_t)bq * 2 + 0];
  float ry = refpts[(size_t)bq * 2 + 1];
  const float* of = offbuf + (size_t)bq * (NHEADS * NLVL * NPTS * 2) + h * (NLVL * NPTS * 2);
  const float Wl[3] = {100.f, 50.f, 25.f};
  const float Hl[3] = {100.f, 50.f, 25.f};
  float* gr = grid + (size_t)t * 48;
#pragma unroll
  for (int l = 0; l < 3; ++l) {
    float wf = Wl[l], hf = Hl[l];
#pragma unroll
    for (int p = 0; p < 8; ++p) {
      float ox = of[(l * 8 + p) * 2 + 0];
      float oy = of[(l * 8 + p) * 2 + 1];
      float lx = rx + ox / wf;                // norm = [w, h]
      float ly = ry + oy / hf;
      // x = ((2*loc-1 + 1)*dim - 1)*0.5 = loc*dim - 0.5  (pixel coords)
      gr[(l * 8 + p) * 2 + 0] = lx * wf - 0.5f;
      gr[(l * 8 + p) * 2 + 1] = ly * hf - 0.5f;
    }
  }
}

// ---------------- Bilinear gather + attention-weighted accumulation ------------
__device__ __forceinline__ float corner(const float* __restrict__ vl, int xi, int yi,
                                        int w, int h, int d) {
  if ((unsigned)xi >= (unsigned)w || (unsigned)yi >= (unsigned)h) return 0.f;
  return vl[((size_t)(yi * w + xi) << 5) + d];   // HD=32 contiguous -> 128B/wave coalesced
}

__global__ void sample_kernel(const float* __restrict__ valueT, const float* __restrict__ attnw,
                              const float* __restrict__ grid, float* __restrict__ mid) {
  int wv = (int)((blockIdx.x * blockDim.x + threadIdx.x) >> 5);  // one wave per (b,q,h)
  int lane = threadIdx.x & 31;                                    // lane = channel d
  if (wv >= BATCH * NQ * NHEADS) return;
  int h = wv & (NHEADS - 1);
  int bq = wv >> 3;
  int b = bq / NQ;
  const float* vbase = valueT + (((size_t)b * NHEADS + h) * SLEN << 5);
  const float* aw = attnw + (size_t)wv * 24;
  const float* gr = grid + (size_t)wv * 48;
  const int Wl[3] = {100, 50, 25};
  const int Hl[3] = {100, 50, 25};
  const int St[3] = {0, 10000, 12500};
  float acc = 0.f;
#pragma unroll
  for (int l = 0; l < 3; ++l) {
    const float* vl = vbase + ((size_t)St[l] << 5);
    int w = Wl[l], hh = Hl[l];
#pragma unroll
    for (int p = 0; p < 8; ++p) {
      int i = l * 8 + p;
      float a = aw[i];
      float x = gr[i * 2 + 0], y = gr[i * 2 + 1];
      float xf = floorf(x), yf = floorf(y);
      int x0 = (int)xf, y0 = (int)yf;
      float fx = x - xf, fy = y - yf;
      float v00 = corner(vl, x0,     y0,     w, hh, lane);
      float v10 = corner(vl, x0 + 1, y0,     w, hh, lane);
      float v01 = corner(vl, x0,     y0 + 1, w, hh, lane);
      float v11 = corner(vl, x0 + 1, y0 + 1, w, hh, lane);
      float bil = v00 * (1.f - fx) * (1.f - fy) + v10 * fx * (1.f - fy)
                + v01 * (1.f - fx) * fy        + v11 * fx * fy;
      acc += a * bil;
    }
  }
  mid[(size_t)bq * CDIM + h * HDIM + lane] = acc;
}

// ---------------- Host-side orchestration --------------------------------------
extern "C" void kernel_launch(void* const* d_in, const int* in_sizes, int n_in,
                              void* d_out, int out_size, void* d_ws, size_t ws_size,
                              hipStream_t stream) {
  (void)in_sizes; (void)n_in; (void)out_size; (void)ws_size;
  const float* query  = (const float*)d_in[0];
  const float* memory = (const float*)d_in[1];
  const float* refpts = (const float*)d_in[2];
  const float* W_off  = (const float*)d_in[3];
  const float* b_off  = (const float*)d_in[4];
  const float* W_attn = (const float*)d_in[5];
  const float* b_attn = (const float*)d_in[6];
  const float* W_val  = (const float*)d_in[7];
  const float* b_val  = (const float*)d_in[8];
  const float* W_out  = (const float*)d_in[9];
  const float* b_out  = (const float*)d_in[10];
  float* out = (float*)d_out;

  char* p = (char*)d_ws;
  auto alloc = [&](size_t bytes) { char* r = p; p += (bytes + 255) & ~(size_t)255; return r; };
  const size_t MQ = (size_t)BATCH * NQ;     // 8192
  const size_t MS = (size_t)BATCH * SLEN;   // 52500

  float*    valueT  = (float*)alloc(MS * CDIM * 4);          // [B][NH][S][HD] f32
  float*    offbuf  = (float*)alloc(MQ * 384 * 4);
  float*    logits  = (float*)alloc(MQ * 192 * 4);
  float*    attnw   = (float*)alloc(MQ * NHEADS * 24 * 4);
  float*    grid    = (float*)alloc(MQ * NHEADS * 48 * 4);
  float*    mid     = (float*)alloc(MQ * CDIM * 4);
  _Float16* WT_val  = (_Float16*)alloc((size_t)CDIM * CDIM * 2);
  _Float16* WT_off  = (_Float16*)alloc((size_t)384 * CDIM * 2);
  _Float16* WT_attn = (_Float16*)alloc((size_t)192 * CDIM * 2);
  _Float16* WT_out  = (_Float16*)alloc((size_t)CDIM * CDIM * 2);
  _Float16* q16     = (_Float16*)alloc(MQ * CDIM * 2);
  _Float16* m16     = (_Float16*)alloc(MS * CDIM * 2);
  _Float16* mid16   = (_Float16*)alloc(MQ * CDIM * 2);

  // Weight transpose->f16 ([N][K]) and activation f32->f16
  wt_transpose_h<<<(CDIM * CDIM + 255) / 256, 256, 0, stream>>>(W_val, WT_val, CDIM, CDIM);
  wt_transpose_h<<<(CDIM * 384 + 255) / 256, 256, 0, stream>>>(W_off, WT_off, CDIM, 384);
  wt_transpose_h<<<(CDIM * 192 + 255) / 256, 256, 0, stream>>>(W_attn, WT_attn, CDIM, 192);
  wt_transpose_h<<<(CDIM * CDIM + 255) / 256, 256, 0, stream>>>(W_out, WT_out, CDIM, CDIM);
  f32_to_f16<<<((int)(MQ * CDIM) + 255) / 256, 256, 0, stream>>>(query, q16, (int)(MQ * CDIM));
  f32_to_f16<<<((int)(MS * CDIM) + 255) / 256, 256, 0, stream>>>(memory, m16, (int)(MS * CDIM));

  auto launch_gemm = [&](const _Float16* A, const _Float16* BT, const float* bias,
                         float* Cf, _Float16* Ch, int M, int N, int mode) {
    int waves = ((M + 15) / 16) * (N / 16);
    int blocks = (waves * 32 + 255) / 256;
    gemm_wmma<<<blocks, 256, 0, stream>>>(A, BT, bias, Cf, Ch, M, N, CDIM, mode);
  };

  launch_gemm(m16, WT_val, b_val, valueT, nullptr, (int)MS, CDIM, 1);   // value proj -> [B][NH][S][HD]
  launch_gemm(q16, WT_off, b_off, offbuf, nullptr, (int)MQ, 384, 0);    // offsets
  launch_gemm(q16, WT_attn, b_attn, logits, nullptr, (int)MQ, 192, 0);  // attn logits

  int nqh = BATCH * NQ * NHEADS;  // 65536
  softmax_grid_kernel<<<(nqh + 255) / 256, 256, 0, stream>>>(offbuf, logits, refpts, attnw, grid);
  sample_kernel<<<(nqh * 32 + 255) / 256, 256, 0, stream>>>(valueT, attnw, grid, mid);
  f32_to_f16<<<((int)(MQ * CDIM) + 255) / 256, 256, 0, stream>>>(mid, mid16, (int)(MQ * CDIM));
  launch_gemm(mid16, WT_out, b_out, out, nullptr, (int)MQ, CDIM, 0);    // output proj
}
